// PointerNetwork_82806969467424
// MI455X (gfx1250) — matrix operand order repaired and checked by reading, not compiled, not run on hardware
//
#include <hip/hip_runtime.h>
#include <math.h>

typedef unsigned short u16;
typedef __attribute__((ext_vector_type(16))) __bf16 v16bf;
typedef __attribute__((ext_vector_type(8)))  float  v8f;

#define B_  64
#define TQ_ 64
#define TP_ 1024
#define H_  1024
#define A_  1024

#define BM 128
#define BN 128
#define KC 32
#define PAD 8
#define NT 8           // 16-col tiles per wave (BN/16)

__device__ __forceinline__ u16 f2bf(float f) {
    unsigned u = __float_as_uint(f);
    u += 0x7FFFu + ((u >> 16) & 1u);   // round-to-nearest-even
    return (u16)(u >> 16);
}

// -------------------------------------------------------------- converts
__global__ void convert_bf16_kernel(const float* __restrict__ src,
                                    u16* __restrict__ dst, int n) {
    int i = blockIdx.x * blockDim.x + threadIdx.x;
    if (i < n) dst[i] = f2bf(src[i]);
}

// W [K,N] fp32 -> Wt [N,K] bf16 (coalesced reads over n)
__global__ void convert_bf16_t_kernel(const float* __restrict__ src,
                                      u16* __restrict__ dst, int K, int N) {
    int k = blockIdx.x;
    int n = blockIdx.y * blockDim.x + threadIdx.x;
    dst[(size_t)n * K + k] = f2bf(src[(size_t)k * N + n]);
}

// ------------------------------------------------- fused attention GEMM
// s_out[row] += sum_n tanh( (Akeys @ Wk)[row,n] + proj[pb,n] ) * wv[n]
// Akeys: [M,K] fp32 (bf16 on the fly); Wt: [N,K] bf16 (pre-transposed Wk).
union FragBF { v16bf v; float4 f4[2]; };

__device__ __forceinline__ void load_bfrag(const u16 (*Bs)[KC + PAD],
                                           FragBF& b, int nt, int lr, int hi) {
    // B fragment: lanes 0-15 col=lr K{0..7,16..23}; lanes 16-31 K{8..15,24..31}
    const int bcol = nt * 16 + lr;
    b.f4[0] = *(const float4*)&Bs[bcol][hi * 8];
    b.f4[1] = *(const float4*)&Bs[bcol][hi * 8 + 16];
}

__device__ __forceinline__ void compute_tile(
    const u16 (*As)[KC + PAD], const u16 (*Bs)[KC + PAD],
    v8f* c, int wave, int lr, int hi)
{
    // A fragment: lanes 0-15 row=lr, K{0..7,16..23}; lanes 16-31 K{8..15,24..31}
    FragBF a;
    const int arow = wave * 16 + lr;
    a.f4[0] = *(const float4*)&As[arow][hi * 8];
    a.f4[1] = *(const float4*)&As[arow][hi * 8 + 16];

    // depth-3 rotating B-fragment pipeline: keep 2 fragments (4 ds loads)
    // in flight past the current WMMA so waits are partial, not dscnt==0.
    FragBF b[3];
    load_bfrag(Bs, b[0], 0, lr, hi);
    load_bfrag(Bs, b[1], 1, lr, hi);
    #pragma unroll
    for (int nt = 0; nt < NT; ++nt) {
        if (nt + 2 < NT) load_bfrag(Bs, b[(nt + 2) % 3], nt + 2, lr, hi);
        c[nt] = __builtin_amdgcn_wmma_f32_16x16x32_bf16(
                    false, a.v, false, b[nt % 3].v, (short)0, c[nt], false, false);
    }
}

__global__ __launch_bounds__(256) void attn_gemm_kernel(
    const float* __restrict__ Akeys,
    const u16*   __restrict__ Wt,
    const float* __restrict__ proj,
    const float* __restrict__ wv,
    float* __restrict__ s_out,
    int M, int K, int N, int pb_shift)   // pb_shift<0 => shared proj
{
    __shared__ alignas(16) u16 As[BM][KC + PAD];   // keys tile, k-major
    __shared__ alignas(16) u16 Bs[BN][KC + PAD];   // weight tile, n-major

    const int tid  = threadIdx.x;
    const int m0   = blockIdx.x * BM;
    const int n0   = blockIdx.y * BN;
    const int wave = tid >> 5;
    const int lane = tid & 31;
    const int lr   = lane & 15;
    const int hi   = lane >> 4;

    v8f c[NT] = {};

    const int ar0 = tid >> 3;           // 0..31  A rows per pass
    const int ac4 = (tid & 7) * 4;      // 0..28  A k-offset (float4)
    const int bn0 = tid >> 3;           // 0..31  B n-rows per pass
    const int bk4 = (tid & 7) * 4;      // 0..28  B k-offset (ushort4)

    float4  pa[4];
    ushort4 pb[4];

    auto load_global = [&](int k0, float4* fa, ushort4* fb) {
        #pragma unroll
        for (int rr = 0; rr < 4; ++rr)
            fa[rr] = *(const float4*)&Akeys[(size_t)(m0 + ar0 + rr * 32) * K + k0 + ac4];
        #pragma unroll
        for (int pp = 0; pp < 4; ++pp)
            fb[pp] = *(const ushort4*)&Wt[(size_t)(n0 + bn0 + pp * 32) * K + k0 + bk4];
    };
    auto store_lds = [&](const float4* fa, const ushort4* fb) {
        #pragma unroll
        for (int rr = 0; rr < 4; ++rr) {
            ushort4 pk;
            pk.x = f2bf(fa[rr].x); pk.y = f2bf(fa[rr].y);
            pk.z = f2bf(fa[rr].z); pk.w = f2bf(fa[rr].w);
            *(ushort4*)&As[ar0 + rr * 32][ac4] = pk;
        }
        #pragma unroll
        for (int pp = 0; pp < 4; ++pp)
            *(ushort4*)&Bs[bn0 + pp * 32][bk4] = fb[pp];
    };

    // prologue: stage tile 0
    load_global(0, pa, pb);
    store_lds(pa, pb);

    for (int k0 = KC; k0 < K; k0 += KC) {
        __syncthreads();
        float4  na[4];
        ushort4 nb[4];
        load_global(k0, na, nb);                 // overlap with WMMA below
        compute_tile(As, Bs, c, wave, lr, hi);
        __syncthreads();
        store_lds(na, nb);
    }
    __syncthreads();
    compute_tile(As, Bs, c, wave, lr, hi);

    // epilogue: tanh + dot with wv, half-wave reduce, atomic accumulate per row
    const int mrow_base = m0 + wave * 16 + hi * 8;
    #pragma unroll
    for (int r = 0; r < 8; ++r) {
        const int row = mrow_base + r;
        const float* pr = proj + (pb_shift >= 0 ? (size_t)(row >> pb_shift) * N : 0);
        float partial = 0.f;
        #pragma unroll
        for (int nt = 0; nt < NT; ++nt) {
            int n = n0 + nt * 16 + lr;
            partial += tanhf(c[nt][r] + pr[n]) * wv[n];
        }
        partial += __shfl_xor(partial, 1, 32);
        partial += __shfl_xor(partial, 2, 32);
        partial += __shfl_xor(partial, 4, 32);
        partial += __shfl_xor(partial, 8, 32);
        if (lr == 0) atomicAdd(&s_out[row], partial);
    }
}

// ---------------------------------------------------- mask + softmax(dim=T)
__global__ __launch_bounds__(256) void mask_softmax_kernel(
    const float* __restrict__ s, const float* __restrict__ mask,
    float* __restrict__ masked_out, float* __restrict__ alpha, int T)
{
    __shared__ float red[256];
    const int b = blockIdx.x, tid = threadIdx.x;
    const float* sb = s + (size_t)b * T;
    const float* mb = mask + (size_t)b * T;
    float lmax = -3.4e38f;
    for (int t = tid; t < T; t += 256) {
        float v = mb[t] > 0.f ? sb[t] : -1e30f;
        if (masked_out) masked_out[(size_t)b * T + t] = v;
        lmax = fmaxf(lmax, v);
    }
    if (!alpha) return;
    red[tid] = lmax; __syncthreads();
    for (int st = 128; st > 0; st >>= 1) {
        if (tid < st) red[tid] = fmaxf(red[tid], red[tid + st]);
        __syncthreads();
    }
    const float gmax = red[0]; __syncthreads();
    float lsum = 0.f;
    for (int t = tid; t < T; t += 256) {
        float v = mb[t] > 0.f ? sb[t] : -1e30f;
        lsum += expf(v - gmax);
    }
    red[tid] = lsum; __syncthreads();
    for (int st = 128; st > 0; st >>= 1) {
        if (tid < st) red[tid] += red[tid + st];
        __syncthreads();
    }
    const float inv = 1.f / red[0];
    for (int t = tid; t < T; t += 256) {
        float v = mb[t] > 0.f ? sb[t] : -1e30f;
        alpha[(size_t)b * T + t] = expf(v - gmax) * inv;
    }
}

// -------------------------------------------- pooled[b,h] = sum_t a[b,t]*k[b,t,h]
__global__ void pool_kernel(const float* __restrict__ keys,
                            const float* __restrict__ alpha,
                            float* __restrict__ out, int T, int H)
{
    const int b = blockIdx.x;
    const int h = blockIdx.y * blockDim.x + threadIdx.x;
    const float* kb = keys + (size_t)b * T * H + h;
    const float* ab = alpha + (size_t)b * T;
    float acc = 0.f;
    for (int t = 0; t < T; ++t) acc += ab[t] * kb[(size_t)t * H];
    out[(size_t)b * H + h] = acc;
}

// ------------------------------- out[b,n] = sum_h x[b,h]*W[h,n]   (W row-major)
__global__ void proj_nt_kernel(const float* __restrict__ x,
                               const float* __restrict__ W,
                               float* __restrict__ out, int Hin, int N)
{
    const int b = blockIdx.x;
    const int n = blockIdx.y * blockDim.x + threadIdx.x;
    const float* xb = x + (size_t)b * Hin;
    float acc = 0.f;
    for (int h = 0; h < Hin; ++h) acc += xb[h] * W[(size_t)h * N + n];
    out[(size_t)b * N + n] = acc;
}

// ------------------ out[b,j] = sum_h x[b,h]*W[j,h] + bias[j]   (GRU gate GEMM)
__global__ void proj_t_kernel(const float* __restrict__ x,
                              const float* __restrict__ W,
                              const float* __restrict__ bias,
                              float* __restrict__ out, int Hin, int N)
{
    const int b = blockIdx.x;
    const int j = blockIdx.y * blockDim.x + threadIdx.x;
    const float4* xr = (const float4*)(x + (size_t)b * Hin);
    const float4* wr = (const float4*)(W + (size_t)j * Hin);
    float acc = 0.f;
    for (int i = 0; i < Hin / 4; ++i) {
        float4 xv = xr[i], wv = wr[i];
        acc += xv.x * wv.x + xv.y * wv.y + xv.z * wv.z + xv.w * wv.w;
    }
    out[(size_t)b * N + j] = acc + bias[j];
}

// --------------------------------------------------------- GRU cell combine
__global__ void gru_kernel(const float* __restrict__ gi,
                           const float* __restrict__ gh,
                           const float* __restrict__ hprev,
                           float* __restrict__ out, int H)
{
    const int i = blockIdx.x * blockDim.x + threadIdx.x;
    const int b = i / H, h = i % H;
    const size_t base = (size_t)b * 3 * H;
    float i_r = gi[base + h], i_z = gi[base + H + h], i_n = gi[base + 2 * H + h];
    float h_r = gh[base + h], h_z = gh[base + H + h], h_n = gh[base + 2 * H + h];
    float r = 1.f / (1.f + expf(-(i_r + h_r)));
    float z = 1.f / (1.f + expf(-(i_z + h_z)));
    float n = tanhf(i_n + r * h_n);
    out[(size_t)b * H + h] = (1.f - z) * n + z * hprev[(size_t)b * H + h];
}

// ======================================================================
extern "C" void kernel_launch(void* const* d_in, const int* in_sizes, int n_in,
                              void* d_out, int out_size, void* d_ws, size_t ws_size,
                              hipStream_t stream)
{
    (void)in_sizes; (void)n_in; (void)out_size; (void)ws_size;
    const float* question = (const float*)d_in[0];
    const float* qmask    = (const float*)d_in[1];
    const float* passage  = (const float*)d_in[2];
    const float* pmask    = (const float*)d_in[3];
    const float* V_q      = (const float*)d_in[4];
    const float* Wk_q     = (const float*)d_in[5];
    const float* Wq_q     = (const float*)d_in[6];
    const float* w_q      = (const float*)d_in[7];
    const float* Wk_p     = (const float*)d_in[8];
    const float* Wq_p     = (const float*)d_in[9];
    const float* w_p      = (const float*)d_in[10];
    const float* W_ih     = (const float*)d_in[11];
    const float* W_hh     = (const float*)d_in[12];
    const float* b_ih     = (const float*)d_in[13];
    const float* b_hh     = (const float*)d_in[14];
    float* out = (float*)d_out;

    char* ws = (char*)d_ws;
    size_t off = 0;
    auto alloc = [&](size_t bytes) -> void* {
        void* p = ws + off;
        off += (bytes + 255) & ~(size_t)255;
        return p;
    };
    u16*   wkq_t   = (u16*)  alloc((size_t)H_ * A_ * 2);   // [A,H] bf16
    u16*   wkp_t   = (u16*)  alloc((size_t)H_ * A_ * 2);   // [A,H] bf16
    float* vqproj  = (float*)alloc((size_t)A_ * 4);
    float* s_q     = (float*)alloc((size_t)B_ * TQ_ * 4);
    float* alpha_q = (float*)alloc((size_t)B_ * TQ_ * 4);
    float* hidden  = (float*)alloc((size_t)B_ * H_ * 4);
    float* hproj   = (float*)alloc((size_t)B_ * A_ * 4);
    float* s_p     = (float*)alloc((size_t)B_ * TP_ * 4);
    float* alpha_p = (float*)alloc((size_t)B_ * TP_ * 4);
    float* pooled  = (float*)alloc((size_t)B_ * H_ * 4);
    float* gi      = (float*)alloc((size_t)B_ * 3 * H_ * 4);
    float* gh      = (float*)alloc((size_t)B_ * 3 * H_ * 4);
    float* gout    = (float*)alloc((size_t)B_ * H_ * 4);

    // transpose+convert attention key weights: [H,A] fp32 -> [A,H] bf16
    convert_bf16_t_kernel<<<dim3(H_, A_ / 256), 256, 0, stream>>>(Wk_q, wkq_t, H_, A_);
    convert_bf16_t_kernel<<<dim3(H_, A_ / 256), 256, 0, stream>>>(Wk_p, wkp_t, H_, A_);

    // vq_proj = V_q @ Wq_q
    proj_nt_kernel<<<dim3(1, A_ / 256), 256, 0, stream>>>(V_q, Wq_q, vqproj, H_, A_);

    // question pooling -> hidden
    hipMemsetAsync(s_q, 0, (size_t)B_ * TQ_ * 4, stream);
    attn_gemm_kernel<<<dim3((B_ * TQ_) / BM, A_ / BN), 256, 0, stream>>>(
        question, wkq_t, vqproj, w_q, s_q, B_ * TQ_, H_, A_, -1);
    mask_softmax_kernel<<<B_, 256, 0, stream>>>(s_q, qmask, nullptr, alpha_q, TQ_);
    pool_kernel<<<dim3(B_, H_ / 256), 256, 0, stream>>>(question, alpha_q, hidden, TQ_, H_);

    // passage pooling with hidden -> ans_begin + inputs
    proj_nt_kernel<<<dim3(B_, A_ / 256), 256, 0, stream>>>(hidden, Wq_p, hproj, H_, A_);
    hipMemsetAsync(s_p, 0, (size_t)B_ * TP_ * 4, stream);
    attn_gemm_kernel<<<dim3((B_ * TP_) / BM, A_ / BN), 256, 0, stream>>>(
        passage, wkp_t, hproj, w_p, s_p, B_ * TP_, H_, A_, 10);
    mask_softmax_kernel<<<B_, 256, 0, stream>>>(s_p, pmask, out, alpha_p, TP_);
    pool_kernel<<<dim3(B_, H_ / 256), 256, 0, stream>>>(passage, alpha_p, pooled, TP_, H_);

    // GRU step
    proj_t_kernel<<<dim3(B_, (3 * H_) / 256), 256, 0, stream>>>(pooled, W_ih, b_ih, gi, H_, 3 * H_);
    proj_t_kernel<<<dim3(B_, (3 * H_) / 256), 256, 0, stream>>>(hidden, W_hh, b_hh, gh, H_, 3 * H_);
    gru_kernel<<<(B_ * H_) / 256, 256, 0, stream>>>(gi, gh, hidden, gout, H_);

    // passage pooling with GRU output -> ans_end
    proj_nt_kernel<<<dim3(B_, A_ / 256), 256, 0, stream>>>(gout, Wq_p, hproj, H_, A_);
    hipMemsetAsync(s_p, 0, (size_t)B_ * TP_ * 4, stream);
    attn_gemm_kernel<<<dim3((B_ * TP_) / BM, A_ / BN), 256, 0, stream>>>(
        passage, wkp_t, hproj, w_p, s_p, B_ * TP_, H_, A_, 10);
    mask_softmax_kernel<<<B_, 256, 0, stream>>>(s_p, pmask, out + (size_t)B_ * TP_, nullptr, TP_);
}
